// LinearCRF_13408887898658
// MI455X (gfx1250) — matrix operand order repaired and checked by reading, not compiled, not run on hardware
//
#include <hip/hip_runtime.h>

typedef __attribute__((ext_vector_type(16))) _Float16 v16h;
typedef __attribute__((ext_vector_type(8)))  _Float16 v8h;
typedef __attribute__((ext_vector_type(8)))  float    v8f;

#define S_LEN 512
#define B_LEN 256
#define T_LEN 128
#define P_STRIDE 136  // halves; 272B row stride -> 16B aligned, dodges bank conflicts

#define LOG2E 1.4426950408889634f
#define LN2   0.6931471805599453f

// ---------------------------------------------------------------------------
// Kernel 1: pack E[i][j] = exp(trans[i][j]) (f16) into a contiguous-k (k = 16*(lane/16) + h) dense
// 16-bit B-matrix (32x16, K x N) per-lane layout so the forward kernel can
// load B fragments with two contiguous 16B loads per lane.
// Entry (jn, kblk, lane): halves h=0..15 hold E[kblk*32 + (lane<16?0:16) + h][jn*16 + lane%16]
// ---------------------------------------------------------------------------
__global__ void crf_build_bpack(const float* __restrict__ trans,
                                _Float16* __restrict__ bpack) {
    int tid = blockIdx.x * blockDim.x + threadIdx.x;  // 0..1023
    if (tid >= 8 * 4 * 32) return;
    int lane = tid & 31;
    int kblk = (tid >> 5) & 3;
    int jn   = tid >> 7;
    int j     = jn * 16 + (lane & 15);
    int kbase = kblk * 32 + ((lane < 16) ? 0 : 16);
    _Float16* dst = bpack + tid * 16;
#pragma unroll
    for (int h = 0; h < 16; ++h)
        dst[h] = (_Float16)expf(trans[(kbase + h) * T_LEN + j]);
}

// ---------------------------------------------------------------------------
// Kernel 2: numerator + mask count, one thread per batch element.
// ---------------------------------------------------------------------------
__global__ void crf_numerator(const float* __restrict__ em,
                              const long long* __restrict__ labels,
                              const unsigned char* __restrict__ mask,
                              const float* __restrict__ starts,
                              const float* __restrict__ trans,
                              const float* __restrict__ ends,
                              float* __restrict__ num, float* __restrict__ cnt) {
    int b = blockIdx.x * blockDim.x + threadIdx.x;
    if (b >= B_LEN) return;
    long long prev = labels[b];
    float acc = starts[(int)prev];
    float c = 0.f;
    for (int s = 0; s < S_LEN; ++s) {
        long long l = labels[(size_t)s * B_LEN + b];
        float mf = mask[(size_t)s * B_LEN + b] ? 1.f : 0.f;
        c += mf;
        acc += em[((size_t)s * B_LEN + b) * T_LEN + (int)l] * mf;
        if (s > 0) acc += trans[(int)prev * T_LEN + (int)l] * mf;
        prev = l;
    }
    int seq_end = (int)c - 1;
    if (seq_end < 0) seq_end = 0;
    acc += ends[(int)labels[(size_t)seq_end * B_LEN + b]];
    num[b] = acc;
    cnt[b] = c;
}

// ---------------------------------------------------------------------------
// Kernel 3: forward scan. Grid = 16 blocks (16 batch rows each, one WGP each),
// block = 256 threads = 8 waves; wave w owns output N-tile jn = w.
// alpha tile held in registers in the WMMA C layout (elem r <-> row rbase+r).
// ---------------------------------------------------------------------------
__launch_bounds__(256)
__global__ void crf_forward(const float* __restrict__ em,
                            const unsigned char* __restrict__ mask,
                            const float* __restrict__ starts,
                            const float* __restrict__ ends,
                            const _Float16* __restrict__ bpack,
                            float* __restrict__ den) {
    __shared__ __attribute__((aligned(16))) _Float16 Plds[16 * P_STRIDE];
    __shared__ float red[8][16];
    __shared__ float mfin[16];

    const int lane  = threadIdx.x & 31;
    const int wave  = threadIdx.x >> 5;   // == jn
    const int jn    = wave;
    const int col   = lane & 15;
    const int hi    = lane >> 4;          // lane group
    const int rbase = hi * 8;             // C-layout row base
    const int b0    = blockIdx.x * 16;
    const int j     = jn * 16 + col;

    // Constant B fragments (exp(trans) slice for this jn), 32 VGPRs, loaded once.
    v16h Bf[4];
#pragma unroll
    for (int kb = 0; kb < 4; ++kb) {
        const v8h* src = (const v8h*)(bpack + ((size_t)((jn * 4 + kb) * 32 + lane) * 16));
        union { v16h v; v8h h[2]; } u;
        u.h[0] = src[0];
        u.h[1] = src[1];
        Bf[kb] = u.v;
    }

    // alpha0 = starts + emissions[0]
    float alpha[8];
    const float st = starts[j];
#pragma unroll
    for (int r = 0; r < 8; ++r)
        alpha[r] = st + em[((size_t)(b0 + rbase + r)) * T_LEN + j];

    // software-pipelined emissions
    float em_cur[8];
#pragma unroll
    for (int r = 0; r < 8; ++r)
        em_cur[r] = em[((size_t)B_LEN + (b0 + rbase + r)) * T_LEN + j];

    for (int s = 1; s < S_LEN; ++s) {
        // prefetch next step's emissions (hidden behind this step's compute)
        float em_nxt[8];
        if (s + 1 < S_LEN) {
#pragma unroll
            for (int r = 0; r < 8; ++r)
                em_nxt[r] = em[((size_t)(s + 1) * B_LEN + (b0 + rbase + r)) * T_LEN + j];
        } else {
#pragma unroll
            for (int r = 0; r < 8; ++r) em_nxt[r] = 0.f;
        }
        // mask bytes for this lane-group's 8 rows: one aligned 8-byte load
        const uint2 mb2 = *(const uint2*)(mask + (size_t)s * B_LEN + b0 + rbase);

        // per-wave row-max partials (reduce across 16-lane group per VGPR)
#pragma unroll
        for (int r = 0; r < 8; ++r) {
            float v = alpha[r];
            v = fmaxf(v, __shfl_xor(v, 1, 16));
            v = fmaxf(v, __shfl_xor(v, 2, 16));
            v = fmaxf(v, __shfl_xor(v, 4, 16));
            v = fmaxf(v, __shfl_xor(v, 8, 16));
            if (col == 0) red[wave][rbase + r] = v;
        }
        __syncthreads();
        if (wave == 0 && lane < 16) {
            float v = red[0][lane];
#pragma unroll
            for (int w = 1; w < 8; ++w) v = fmaxf(v, red[w][lane]);
            mfin[lane] = v;
        }
        __syncthreads();
        float m[8];
#pragma unroll
        for (int r = 0; r < 8; ++r) m[r] = mfin[rbase + r];

        // P = 2^((alpha - m)*log2e) -> f16 in LDS (raw v_exp_f32; arg <= 0, safe)
#pragma unroll
        for (int r = 0; r < 8; ++r)
            Plds[(rbase + r) * P_STRIDE + j] =
                (_Float16)__builtin_amdgcn_exp2f((alpha[r] - m[r]) * LOG2E);
        __syncthreads();

        // C = P @ E tile via 4 chained WMMAs
        v8f c = {0.f, 0.f, 0.f, 0.f, 0.f, 0.f, 0.f, 0.f};
#pragma unroll
        for (int kb = 0; kb < 4; ++kb) {
            const int row  = lane & 15;
            const int koff = kb * 32 + hi * 16;
            union { v16h v; v8h h[2]; } a;
            a.h[0] = *(const v8h*)(Plds + row * P_STRIDE + koff);
            a.h[1] = *(const v8h*)(Plds + row * P_STRIDE + koff + 8);
            c = __builtin_amdgcn_wmma_f32_16x16x32_f16(
                    false, a.v, false, Bf[kb], (short)0, c, false, false);
        }

        // alpha' = ln2*log2(C) + m + em ; branchless masked blend
        // (C in [0.9, ~141] -> raw v_log_f32 is safe)
#pragma unroll
        for (int r = 0; r < 8; ++r) {
            float anew = fmaf(LN2, __builtin_amdgcn_logf(c[r]), m[r] + em_cur[r]);
            unsigned w  = (r < 4) ? mb2.x : mb2.y;
            float mf    = (float)((w >> (8 * (r & 3))) & 1u);
            alpha[r]   += mf * (anew - alpha[r]);
        }
#pragma unroll
        for (int r = 0; r < 8; ++r) em_cur[r] = em_nxt[r];
    }

    // denominator[b] = logsumexp_j(alpha[b,j] + ends[j])
    float x[8];
    const float e = ends[j];
#pragma unroll
    for (int r = 0; r < 8; ++r) x[r] = alpha[r] + e;
#pragma unroll
    for (int r = 0; r < 8; ++r) {
        float v = x[r];
        v = fmaxf(v, __shfl_xor(v, 1, 16));
        v = fmaxf(v, __shfl_xor(v, 2, 16));
        v = fmaxf(v, __shfl_xor(v, 4, 16));
        v = fmaxf(v, __shfl_xor(v, 8, 16));
        if (col == 0) red[wave][rbase + r] = v;
    }
    __syncthreads();
    if (wave == 0 && lane < 16) {
        float v = red[0][lane];
#pragma unroll
        for (int w = 1; w < 8; ++w) v = fmaxf(v, red[w][lane]);
        mfin[lane] = v;
    }
    __syncthreads();
#pragma unroll
    for (int r = 0; r < 8; ++r) {
        float v = __builtin_amdgcn_exp2f((x[r] - mfin[rbase + r]) * LOG2E);
        v += __shfl_xor(v, 1, 16);
        v += __shfl_xor(v, 2, 16);
        v += __shfl_xor(v, 4, 16);
        v += __shfl_xor(v, 8, 16);
        x[r] = v;  // per-row partial sum (this wave's 16 columns)
    }
    __syncthreads();
    if (col == 0) {
#pragma unroll
        for (int r = 0; r < 8; ++r) red[wave][rbase + r] = x[r];
    }
    __syncthreads();
    if (wave == 0 && lane < 16) {
        float v = 0.f;
#pragma unroll
        for (int w = 0; w < 8; ++w) v += red[w][lane];
        den[b0 + lane] = fmaf(LN2, __builtin_amdgcn_logf(v), mfin[lane]);
    }
}

// ---------------------------------------------------------------------------
// Kernel 4: final reduction -> scalar output
// ---------------------------------------------------------------------------
__global__ void crf_reduce(const float* __restrict__ num,
                           const float* __restrict__ den,
                           const float* __restrict__ cnt,
                           float* __restrict__ out) {
    __shared__ float sn[256], sd[256], sc[256];
    int t = threadIdx.x;
    sn[t] = num[t]; sd[t] = den[t]; sc[t] = cnt[t];
    __syncthreads();
    for (int off = 128; off > 0; off >>= 1) {
        if (t < off) { sn[t] += sn[t + off]; sd[t] += sd[t + off]; sc[t] += sc[t + off]; }
        __syncthreads();
    }
    if (t == 0) out[0] = -(sn[0] - sd[0]) / sc[0];
}

extern "C" void kernel_launch(void* const* d_in, const int* in_sizes, int n_in,
                              void* d_out, int out_size, void* d_ws, size_t ws_size,
                              hipStream_t stream) {
    const float*         em     = (const float*)d_in[0];
    const long long*     labels = (const long long*)d_in[1];     // int64
    const unsigned char* mask   = (const unsigned char*)d_in[2]; // bool
    const float*         starts = (const float*)d_in[3];
    const float*         trans  = (const float*)d_in[4];
    const float*         ends   = (const float*)d_in[5];

    _Float16* bpack = (_Float16*)d_ws;                 // 32 KB
    float* num = (float*)((char*)d_ws + 32768);        // 1 KB
    float* den = num + B_LEN;                          // 1 KB
    float* cnt = den + B_LEN;                          // 1 KB
    float* out = (float*)d_out;

    crf_build_bpack<<<4, 256, 0, stream>>>(trans, bpack);
    crf_numerator<<<1, 256, 0, stream>>>(em, labels, mask, starts, trans, ends, num, cnt);
    crf_forward<<<16, 256, 0, stream>>>(em, mask, starts, ends, bpack, den);
    crf_reduce<<<1, 256, 0, stream>>>(num, den, cnt, out);
}
